// HybridResNetQNN_6485400617737
// MI455X (gfx1250) — matrix operand order repaired and checked by reading, not compiled, not run on hardware
//
#include <hip/hip_runtime.h>
#include <math.h>

typedef __attribute__((ext_vector_type(16))) _Float16 v16h;
typedef __attribute__((ext_vector_type(8)))  _Float16 v8h;
typedef __attribute__((ext_vector_type(8)))  float    v8f;

// ---------------- pack x: NCHW f32 -> NHWC f16 with channel zero-pad to Cpad ----------------
__global__ void k_pack_x_pad(const float* __restrict__ in, _Float16* __restrict__ out,
                             int N, int C, int Cpad, int H, int W)
{
  int i = blockIdx.x * blockDim.x + threadIdx.x;
  int total = N * H * W * Cpad;
  if (i >= total) return;
  int c = i % Cpad; int t = i / Cpad;
  int x = t % W; t /= W;
  int y = t % H; int b = t / H;
  float v = (c < C) ? in[((b * C + c) * H + y) * W + x] : 0.0f;
  out[i] = (_Float16)v;
}

// ---------------- pack weights: OIHW f32 -> O,KH,KW,Ipad f16 (zero-pad input channels) ----------------
__global__ void k_pack_w_ohwi(const float* __restrict__ in, _Float16* __restrict__ out,
                              int O, int Cin, int Cpad, int KH, int KW)
{
  int i = blockIdx.x * blockDim.x + threadIdx.x;
  int total = O * KH * KW * Cpad;
  if (i >= total) return;
  int c  = i % Cpad; int t = i / Cpad;
  int kx = t % KW;   t /= KW;
  int ky = t % KH;   int o = t / KH;
  float v = (c < Cin) ? in[((o * Cin + c) * KH + ky) * KW + kx] : 0.0f;
  out[i] = (_Float16)v;
}

// ================= fast NHWC implicit-GEMM conv (Cin%32==0, Cout%64==0) =================
// One wave computes a 64(Cout) x 16(N*Ho*Wo) tile: 4 M-sub-tiles share one B fragment.
// A = OHWI weights, B = NHWC input; per filter tap the K-slice over cin is contiguous
// in BOTH operands -> pure b128 loads, 4 independent-accumulator WMMAs per B load.
template <int KH, int KW>
__global__ __launch_bounds__(256) void k_conv_nhwc_fast(
    const _Float16* __restrict__ in, const _Float16* __restrict__ wt,
    const float* __restrict__ bnb, const float* __restrict__ bng,
    const float* __restrict__ bnm, const float* __restrict__ bnv,
    const _Float16* __restrict__ res, _Float16* __restrict__ out,
    int N, int Cin, int H, int W, int Cout, int Ho, int Wo,
    int stride, int pad, int relu)
{
  const int lane = threadIdx.x & 31;
  const int wave = blockIdx.x * (blockDim.x >> 5) + (threadIdx.x >> 5);
  const int nTilesN = (N * Ho * Wo) >> 4;          // exact: N=16
  if (wave >= (Cout >> 6) * nTilesN) return;       // wave-uniform
  const int mb = wave / nTilesN;                   // 64-channel M block
  const int nt = wave - mb * nTilesN;

  const int col = lane & 15;                       // A row / B col / C col
  const int hi  = lane >> 4;
  const int kbA = hi << 3;                         // A: K 0-7,16-23 | 8-15,24-31
  const int kbB = hi << 3;                         // B: K 0-7,16-23 | 8-15,24-31

  const int HoWo = Ho * Wo;
  const int s  = nt * 16 + col;
  const int bb = s / HoWo;
  const int rm = s - bb * HoWo;
  const int oy = rm / Wo;
  const int ox = rm - oy * Wo;
  const int coutA = mb * 64 + col;                 // A row of sub-tile 0

  v8f acc0 = {}, acc1 = {}, acc2 = {}, acc3 = {};
#pragma unroll
  for (int ky = 0; ky < KH; ++ky) {
    const int iy = oy * stride - pad + ky;
    const bool rowOK = ((unsigned)iy < (unsigned)H);
#pragma unroll
    for (int kx = 0; kx < KW; ++kx) {
      const int ix = ox * stride - pad + kx;
      const bool ok = rowOK && ((unsigned)ix < (unsigned)W);
      const _Float16* wtp = wt + (size_t)((coutA * KH + ky) * KW + kx) * Cin;
      const _Float16* inp = in + (size_t)((bb * H + iy) * W + ix) * Cin;
      const size_t wsub = (size_t)16 * KH * KW * Cin;   // A stride between sub-tiles
      for (int c0 = 0; c0 < Cin; c0 += 32) {
        // ---- one B fragment, shared by 4 WMMAs ----
        v16h bf;
        if (ok) {
          const v8h blo = *(const v8h*)(inp + c0 + kbB);
          const v8h bhi = *(const v8h*)(inp + c0 + kbB + 16);
#pragma unroll
          for (int j = 0; j < 8; ++j) { bf[j] = blo[j]; bf[j + 8] = bhi[j]; }
        } else {
#pragma unroll
          for (int j = 0; j < 16; ++j) bf[j] = (_Float16)0.0f;
        }
        // ---- 4 A fragments (sub-tiles of 16 output channels each) ----
        v16h af0, af1, af2, af3;
#pragma unroll
        for (int sub = 0; sub < 4; ++sub) {
          const _Float16* wp = wtp + (size_t)sub * wsub + c0;
          const v8h alo = *(const v8h*)(wp + kbA);
          const v8h ahi = *(const v8h*)(wp + kbA + 16);
          v16h af;
#pragma unroll
          for (int j = 0; j < 8; ++j) { af[j] = alo[j]; af[j + 8] = ahi[j]; }
          if (sub == 0) af0 = af; else if (sub == 1) af1 = af;
          else if (sub == 2) af2 = af; else af3 = af;
        }
        acc0 = __builtin_amdgcn_wmma_f32_16x16x32_f16(false, af0, false, bf, (short)0, acc0, false, false);
        acc1 = __builtin_amdgcn_wmma_f32_16x16x32_f16(false, af1, false, bf, (short)0, acc1, false, false);
        acc2 = __builtin_amdgcn_wmma_f32_16x16x32_f16(false, af2, false, bf, (short)0, acc2, false, false);
        acc3 = __builtin_amdgcn_wmma_f32_16x16x32_f16(false, af3, false, bf, (short)0, acc3, false, false);
      }
    }
  }

  // epilogue: 4 sub-tiles x 8 contiguous channels per lane -> fused BN/res/ReLU + b128 stores
  const size_t spat = (size_t)((bb * Ho + oy) * Wo + ox) * Cout;
#pragma unroll
  for (int sub = 0; sub < 4; ++sub) {
    const v8f acc = (sub == 0) ? acc0 : (sub == 1) ? acc1 : (sub == 2) ? acc2 : acc3;
    const int coutBase = mb * 64 + sub * 16 + (hi << 3);
    const size_t obase = spat + coutBase;
    float vv[8];
#pragma unroll
    for (int r = 0; r < 8; ++r) {
      int cout = coutBase + r;
      float sc = bng[cout] * rsqrtf(bnv[cout] + 1e-5f);
      vv[r] = (acc[r] - bnm[cout]) * sc + bnb[cout];
    }
    if (res) {
      const v8h rv = *(const v8h*)(res + obase);
#pragma unroll
      for (int r = 0; r < 8; ++r) vv[r] += (float)rv[r];
    }
    v8h ov;
#pragma unroll
    for (int r = 0; r < 8; ++r) {
      float v = vv[r];
      if (relu) v = v > 0.0f ? v : 0.0f;
      ov[r] = (_Float16)v;
    }
    *(v8h*)(out + obase) = ov;
  }
}

// ---------------- 3x3 s2 p1 maxpool (NHWC f16) ----------------
__global__ void k_maxpool3s2(const _Float16* __restrict__ in, _Float16* __restrict__ out,
                             int N, int C, int H, int W, int Ho, int Wo)
{
  int i = blockIdx.x * blockDim.x + threadIdx.x;
  int total = N * Ho * Wo * C;
  if (i >= total) return;
  int c  = i % C;  int t = i / C;
  int ox = t % Wo; t /= Wo;
  int oy = t % Ho; int b = t / Ho;
  float m = -1e30f;
  for (int ky = 0; ky < 3; ++ky)
    for (int kx = 0; kx < 3; ++kx) {
      int iy = oy * 2 - 1 + ky, ix = ox * 2 - 1 + kx;
      if ((unsigned)iy < (unsigned)H && (unsigned)ix < (unsigned)W) {
        float v = (float)in[(size_t)((b * H + iy) * W + ix) * C + c];
        m = v > m ? v : m;
      }
    }
  out[i] = (_Float16)m;
}

// ---------------- global average pool (NHWC f16 -> f32 feats[b][512]) ----------------
__global__ void k_avgpool(const _Float16* __restrict__ in, float* __restrict__ feats,
                          int n, int C, int HW)
{
  int i = blockIdx.x * blockDim.x + threadIdx.x;
  if (i >= n) return;
  int c = i % C, b = i / C;
  float s = 0.f;
  for (int k = 0; k < HW; ++k) s += (float)in[(size_t)(b * HW + k) * C + c];
  feats[i] = s / (float)HW;
}

// ---------------- head: proj+ReLU, 4-qubit statevector QNN, classifier ----------------
__global__ void k_head(const float* __restrict__ feats, const float* __restrict__ proj_w,
                       const float* __restrict__ proj_b, const float* __restrict__ qw,
                       const float* __restrict__ cls_w, const float* __restrict__ cls_b,
                       float* __restrict__ out)
{
  int b = threadIdx.x;
  if (b >= 16) return;
  float h[4];
  for (int p = 0; p < 4; ++p) {
    float a = proj_b[p];
    for (int k = 0; k < 512; ++k) a += feats[b * 512 + k] * proj_w[p * 512 + k];
    h[p] = a > 0.f ? a : 0.f;
  }
  float cs[4], sn[4];
  for (int w = 0; w < 4; ++w) { cs[w] = cosf(h[w] * 0.5f); sn[w] = sinf(h[w] * 0.5f); }
  float re[16], im[16];
  for (int idx = 0; idx < 16; ++idx) {
    float mag = 1.f; int n1 = 0;
    for (int w = 0; w < 4; ++w) {
      int bit = (idx >> (3 - w)) & 1;
      mag *= bit ? sn[w] : cs[w];
      n1 += bit;
    }
    float rr = 0.f, ii = 0.f;             // (-i)^n1 * mag
    switch (n1 & 3) { case 0: rr = mag; break; case 1: ii = -mag; break;
                      case 2: rr = -mag; break; default: ii = mag; }
    re[idx] = rr; im[idx] = ii;
  }
  for (int l = 0; l < 4; ++l) {
    for (int w = 0; w < 4; ++w) {
      float phi = qw[l * 12 + w * 3 + 0];
      float th  = qw[l * 12 + w * 3 + 1];
      float om  = qw[l * 12 + w * 3 + 2];
      float ct = cosf(th * 0.5f), st = sinf(th * 0.5f);
      float a1 = 0.5f * (phi + om), a2 = 0.5f * (phi - om);
      float c1 = cosf(a1), s1 = sinf(a1), c2 = cosf(a2), s2 = sinf(a2);
      float u00r =  c1 * ct, u00i = -s1 * ct;
      float u01r = -c2 * st, u01i = -s2 * st;
      float u10r =  c2 * st, u10i = -s2 * st;
      float u11r =  c1 * ct, u11i =  s1 * ct;
      int mask = 8 >> w;
      for (int i0 = 0; i0 < 16; ++i0) {
        if (i0 & mask) continue;
        int i1 = i0 | mask;
        float r0 = re[i0], m0 = im[i0], r1 = re[i1], m1 = im[i1];
        re[i0] = u00r * r0 - u00i * m0 + u01r * r1 - u01i * m1;
        im[i0] = u00r * m0 + u00i * r0 + u01r * m1 + u01i * r1;
        re[i1] = u10r * r0 - u10i * m0 + u11r * r1 - u11i * m1;
        im[i1] = u10r * m0 + u10i * r0 + u11r * m1 + u11i * r1;
      }
    }
    int r = (l % 3) + 1;
    for (int i = 0; i < 4; ++i) {
      int cmask = 8 >> i, tmask = 8 >> ((i + r) % 4);
      for (int idx = 0; idx < 16; ++idx) {
        if ((idx & cmask) && !(idx & tmask)) {
          int j = idx | tmask;
          float tr = re[idx]; re[idx] = re[j]; re[j] = tr;
          float ti = im[idx]; im[idx] = im[j]; im[j] = ti;
        }
      }
    }
  }
  float ev = 0.f;
  for (int i = 0; i < 16; ++i) {
    float p2 = re[i] * re[i] + im[i] * im[i];
    ev += (i & 8) ? -p2 : p2;
  }
  out[b * 2 + 0] = ev * cls_w[0] + cls_b[0];
  out[b * 2 + 1] = ev * cls_w[1] + cls_b[1];
}

// ================= host orchestration =================
extern "C" void kernel_launch(void* const* d_in, const int* in_sizes, int n_in,
                              void* d_out, int out_size, void* d_ws, size_t ws_size,
                              hipStream_t stream)
{
  (void)in_sizes; (void)n_in; (void)out_size; (void)ws_size;
  struct Bn { const float *b, *g, *m, *v; };

  // jax tree-flatten order (sorted dict keys): d_in[0]=x, then params leaves.
  int ci = 1;
  auto nxt   = [&]() { return (const float*)d_in[ci++]; };
  auto nxtBn = [&]() { Bn r; r.b = nxt(); r.g = nxt(); r.m = nxt(); r.v = nxt(); return r; };

  Bn bbn1 = nxtBn();
  const float* w_conv1 = nxt();

  struct Blk { Bn bn1, bn2, dbn; const float *w1, *w2, *dw; };
  Blk blk[8] = {};
  const bool downs[8] = {false,false, true,false, true,false, true,false};
  for (int i = 0; i < 8; ++i) {
    blk[i].bn1 = nxtBn(); blk[i].bn2 = nxtBn();
    blk[i].w1  = nxt();   blk[i].w2  = nxt();
    blk[i].dw  = nullptr;
    if (downs[i]) { blk[i].dbn = nxtBn(); blk[i].dw = nxt(); }
  }
  const float* cls_b  = nxt();
  const float* cls_w  = nxt();
  const float* proj_b = nxt();
  const float* proj_w = nxt();
  const float* qw     = nxt();

  // workspace carve-out
  size_t off = 0;
  auto alloc = [&](size_t bytes) -> void* {
    void* p = (char*)d_ws + off;
    off += (bytes + 255) & ~(size_t)255;
    return p;
  };
  auto packW = [&](const float* src, int O, int I, int Ipad, int K) -> const _Float16* {
    int n = O * K * K * Ipad;
    _Float16* dst = (_Float16*)alloc((size_t)n * sizeof(_Float16));
    k_pack_w_ohwi<<<(n + 255) / 256, 256, 0, stream>>>(src, dst, O, I, Ipad, K, K);
    return dst;
  };

  const int N = 16;
  const int cins[8]  = {64,64, 64,128, 128,256, 256,512};
  const int couts[8] = {64,64, 128,128, 256,256, 512,512};
  const int strds[8] = {1,1, 2,1, 2,1, 2,1};

  const _Float16* hw_conv1 = packW(w_conv1, 64, 3, 32, 7);   // stem padded to Cin=32
  const _Float16 *hw1[8], *hw2[8], *hwd[8];
  for (int i = 0; i < 8; ++i) {
    hw1[i] = packW(blk[i].w1, couts[i], cins[i], cins[i], 3);
    hw2[i] = packW(blk[i].w2, couts[i], couts[i], couts[i], 3);
    hwd[i] = downs[i] ? packW(blk[i].dw, couts[i], cins[i], cins[i], 1) : nullptr;
  }

  const size_t ACT = (size_t)N * 64 * 112 * 112;   // max activation elements (25.7 MB f16)
  _Float16* bufA = (_Float16*)alloc(ACT * 2);
  _Float16* bufB = (_Float16*)alloc(ACT * 2);
  _Float16* bufC = (_Float16*)alloc(ACT * 2);      // contiguous with bufS below
  _Float16* bufS = (_Float16*)alloc(ACT * 2);
  float* feats = (float*)alloc((size_t)N * 512 * sizeof(float));
  // padded stem input (16x224x224x32 = 2*ACT elems) aliases bufC+bufS: it is fully
  // consumed by the stem conv before bufC/bufS are first written (first block).
  _Float16* xpad = bufC;

  auto conv = [&](const _Float16* in, const _Float16* w, Bn bn, const _Float16* res,
                  _Float16* outp, int Cin, int Hh, int Ww, int Cout,
                  int KH, int s, int p, int relu) {
    int Ho = (Hh + 2 * p - KH) / s + 1;
    int Wo = (Ww + 2 * p - KH) / s + 1;
    int waves = (Cout / 64) * ((N * Ho * Wo) / 16);
    int blocks = (waves + 7) / 8;                  // 256 threads = 8 waves/block
    if (KH == 3)
      k_conv_nhwc_fast<3,3><<<blocks, 256, 0, stream>>>(in, w, bn.b, bn.g, bn.m, bn.v,
          res, outp, N, Cin, Hh, Ww, Cout, Ho, Wo, s, p, relu);
    else if (KH == 1)
      k_conv_nhwc_fast<1,1><<<blocks, 256, 0, stream>>>(in, w, bn.b, bn.g, bn.m, bn.v,
          res, outp, N, Cin, Hh, Ww, Cout, Ho, Wo, s, p, relu);
    else
      k_conv_nhwc_fast<7,7><<<blocks, 256, 0, stream>>>(in, w, bn.b, bn.g, bn.m, bn.v,
          res, outp, N, Cin, Hh, Ww, Cout, Ho, Wo, s, p, relu);
  };

  // ---- stem: pack x (NCHW f32 -> NHWC f16, C 3->32), conv7x7 s2 + bn + relu, maxpool ----
  {
    int n = N * 224 * 224 * 32;
    k_pack_x_pad<<<(n + 255) / 256, 256, 0, stream>>>((const float*)d_in[0], xpad,
                                                      N, 3, 32, 224, 224);
  }
  conv(xpad, hw_conv1, bbn1, nullptr, bufB, 32, 224, 224, 64, 7, 2, 3, 1);
  {
    int total = N * 56 * 56 * 64;
    k_maxpool3s2<<<(total + 255) / 256, 256, 0, stream>>>(bufB, bufA, N, 64, 112, 112, 56, 56);
  }

  // ---- residual blocks ----
  _Float16* cur = bufA;
  _Float16* t1  = bufB;
  _Float16* t2  = bufC;
  int Hc = 56;
  for (int i = 0; i < 8; ++i) {
    int Hout = Hc / strds[i];
    const _Float16* resp;
    if (downs[i]) {
      conv(cur, hwd[i], blk[i].dbn, nullptr, bufS, cins[i], Hc, Hc, couts[i], 1, strds[i], 0, 0);
      resp = bufS;
    } else {
      resp = cur;
    }
    conv(cur, hw1[i], blk[i].bn1, nullptr, t1, cins[i], Hc, Hc, couts[i], 3, strds[i], 1, 1);
    conv(t1,  hw2[i], blk[i].bn2, resp,   t2, couts[i], Hout, Hout, couts[i], 3, 1, 1, 1);
    _Float16* tmp = cur; cur = t2; t2 = tmp;       // ping-pong
    Hc = Hout;
  }

  // ---- global average pool (NHWC 16x7x7x512 -> 16x512) ----
  {
    int n = N * 512;
    k_avgpool<<<(n + 255) / 256, 256, 0, stream>>>(cur, feats, n, 512, 49);
  }
  // ---- projection + QNN + classifier ----
  k_head<<<1, 32, 0, stream>>>(feats, proj_w, proj_b, qw, cls_w, cls_b, (float*)d_out);
}
